// MultiHeadAttention_49976239456305
// MI455X (gfx1250) — matrix operand order repaired and checked by reading, not compiled, not run on hardware
//
#include <hip/hip_runtime.h>
#include <hip/hip_bf16.h>
#include <stdint.h>

// ---------------------------------------------------------------------------
// ESM3-style MHA block for MI455X (gfx1250, wave32, WMMA).
// B=2, S=2048, D=1024, H=16, dh=64 (fixed by the reference setup).
// GEMMs: v_wmma_f32_16x16x32_bf16 (K=32/instr, f32 accum).
// Tile staging: GLOBAL_LOAD_ASYNC_TO_LDS_B128 (ASYNCcnt) + global_prefetch_b8.
// Layout rule: contraction dim innermost for every WMMA operand, so each
// 16-element fragment per lane is two contiguous b128 LDS/global loads.
// ---------------------------------------------------------------------------

#define BB 2
#define SS 2048
#define DD 1024
#define HH 16
#define DH 64
#define NT (BB * SS)         // 4096 tokens
#define LN_EPS 1e-5f
#define NEG_BIG (-3.0e38f)

typedef __attribute__((ext_vector_type(16))) __bf16 v16bf;
typedef __attribute__((ext_vector_type(8)))  float  v8f;
typedef __attribute__((address_space(3))) const void lds_cv;

union FragBF { uint4 q[2]; v16bf v; };

__device__ __forceinline__ __bf16 f2bf(float f) {
    union { float f; uint32_t u; } uu; uu.f = f;
    uint32_t r = uu.u + 0x7FFFu + ((uu.u >> 16) & 1u);   // round-nearest-even
    uint16_t h = (uint16_t)(r >> 16);
    __bf16 b; __builtin_memcpy(&b, &h, 2); return b;
}

// gfx1250 async copy: global -> LDS, 16B per lane, tracked by ASYNCcnt.
__device__ __forceinline__ void async_ld_b128(const void* gptr, void* lptr) {
    unsigned loff = (unsigned)(uintptr_t)(lds_cv*)lptr;      // LDS byte offset
    asm volatile("global_load_async_to_lds_b128 %0, %1, off"
                 :: "v"(loff), "v"(gptr) : "memory");
}
__device__ __forceinline__ void wait_async0() {
#if __has_builtin(__builtin_amdgcn_s_wait_asynccnt)
    __builtin_amdgcn_s_wait_asynccnt(0);
#else
    asm volatile("s_wait_asynccnt 0x0" ::: "memory");
#endif
}

// A-operand fragment (16x32 bf16): lane m = lane&15, K split by lane half.
// ISA layout -> two contiguous 8-elem runs at k = kh*8 and k = kh*8 + 16.
__device__ __forceinline__ v16bf load_frag_a(const __bf16* rowptr, int kh) {
    FragBF f;
    f.q[0] = *(const uint4*)(rowptr + kh * 8);
    f.q[1] = *(const uint4*)(rowptr + kh * 8 + 16);
    return f.v;
}
// B-operand fragment (32x16 bf16): lane n = lane&15, same k order as A:
// two contiguous 8-elem runs at k = kh*8 and k = kh*8 + 16.
__device__ __forceinline__ v16bf load_frag_b(const __bf16* colptr, int kh) {
    FragBF f;
    f.q[0] = *(const uint4*)(colptr + kh * 8);
    f.q[1] = *(const uint4*)(colptr + kh * 8 + 16);
    return f.v;
}

// ------------------------- kernel 1: layernorm(x) --------------------------
__global__ __launch_bounds__(256)
void ln_x_kernel(const float* __restrict__ x, const float* __restrict__ w,
                 const float* __restrict__ b, __bf16* __restrict__ out) {
    const int row = blockIdx.x;                 // token
    const float* xr = x + (size_t)row * DD;
    float s = 0.f, s2 = 0.f;
    for (int i = threadIdx.x; i < DD; i += 256) { float v = xr[i]; s += v; s2 += v * v; }
    #pragma unroll
    for (int o = 16; o > 0; o >>= 1) { s += __shfl_down(s, o, 32); s2 += __shfl_down(s2, o, 32); }
    __shared__ float red[2][8];
    const int wv = threadIdx.x >> 5, ln = threadIdx.x & 31;
    if (ln == 0) { red[0][wv] = s; red[1][wv] = s2; }
    __syncthreads();
    float ts = 0.f, ts2 = 0.f;
    #pragma unroll
    for (int i = 0; i < 8; ++i) { ts += red[0][i]; ts2 += red[1][i]; }
    const float mean = ts / DD;
    const float var  = ts2 / DD - mean * mean;
    const float rstd = rsqrtf(var + LN_EPS);
    for (int i = threadIdx.x; i < DD; i += 256)
        out[(size_t)row * DD + i] = f2bf((xr[i] - mean) * rstd * w[i] + b[i]);
}

// ------------- kernel 2: transpose + convert weights (f32->bf16) -----------
// out[n*K + k] = in[k*N + n]; LDS-tiled so both sides stay coalesced.
__global__ __launch_bounds__(256)
void transpose_cvt_kernel(const float* __restrict__ in, __bf16* __restrict__ out,
                          int K, int N) {
    __shared__ float tile[32][33];
    const int kb = blockIdx.y * 32, nb = blockIdx.x * 32;
    const int tx = threadIdx.x & 31, ty = threadIdx.x >> 5;
    #pragma unroll
    for (int r = ty; r < 32; r += 8)
        tile[r][tx] = in[(size_t)(kb + r) * N + nb + tx];
    __syncthreads();
    #pragma unroll
    for (int r = ty; r < 32; r += 8)
        out[(size_t)(nb + r) * K + kb + tx] = f2bf(tile[tx][r]);
}

// ------------------ kernel 3/6: bf16 WMMA GEMM, f32 out --------------------
// C[M,N] = A[M,K] * Bt[N,K]^T  (Bt row-major = B transposed), M%128, N%64, K%32.
// Block = 256 threads = 8 waves; macro-tile 128x64; wave w -> rows 16w..16w+15,
// 4 output tiles of 16x16 along N, K staged in LDS 32 at a time via async DMA.
__global__ __launch_bounds__(256)
void gemm_bf16_kernel(const __bf16* __restrict__ A, const __bf16* __restrict__ Bt,
                      float* __restrict__ C, int M, int N, int K) {
    __shared__ alignas(16) __bf16 sA[128 * 32];   // [row][k]
    __shared__ alignas(16) __bf16 sBt[64 * 32];   // [col][k]  (contraction innermost)
    const int tid  = threadIdx.x;
    const int wv   = tid >> 5;
    const int lane = tid & 31;
    const int l15  = lane & 15;
    const int kh   = lane >> 4;
    const int row0 = blockIdx.y * 128;
    const int col0 = blockIdx.x * 64;

    // per-thread staging coordinates (branch-free loop body)
    const int ar = tid >> 1, ac = (tid & 1) * 16;          // A: 16 elems/thread
    const int bn = tid >> 2, bc = (tid & 3) * 8;           // Bt: 8 elems/thread
    const __bf16* gA = A  + (size_t)(row0 + ar) * K + ac;
    const __bf16* gB = Bt + (size_t)(col0 + bn) * K + bc;
    __bf16* lA = &sA[ar * 32 + ac];
    __bf16* lB = &sBt[bn * 32 + bc];

    v8f acc[4];
    #pragma unroll
    for (int t = 0; t < 4; ++t)
        #pragma unroll
        for (int r = 0; r < 8; ++r) acc[t][r] = 0.f;

    for (int k0 = 0; k0 < K; k0 += 32) {
        // async DMA tile staging (no VGPR round-trip), ASYNCcnt-tracked
        async_ld_b128(gA + k0,      lA);
        async_ld_b128(gA + k0 + 8,  lA + 8);
        async_ld_b128(gB + k0,      lB);
        // speculative prefetch of next K-step (silently dropped past the end)
        __builtin_prefetch(gA + k0 + 32, 0, 1);
        __builtin_prefetch(gB + k0 + 32, 0, 1);
        wait_async0();
        __syncthreads();

        const v16bf a = load_frag_a(&sA[((wv << 4) + l15) * 32], kh);
        #pragma unroll
        for (int nt = 0; nt < 4; ++nt) {
            const v16bf bb = load_frag_b(&sBt[(nt * 16 + l15) * 32], kh);
            acc[nt] = __builtin_amdgcn_wmma_f32_16x16x32_bf16(
                false, a, false, bb, (short)0, acc[nt], false, false);
        }
        __syncthreads();
    }
    // C/D layout: lanes 0-15 -> M=r, lanes 16-31 -> M=r+8; N = lane&15
    #pragma unroll
    for (int nt = 0; nt < 4; ++nt)
        #pragma unroll
        for (int r = 0; r < 8; ++r) {
            const int m = (wv << 4) + r + 8 * kh;
            C[(size_t)(row0 + m) * N + col0 + nt * 16 + l15] = acc[nt][r];
        }
}

// --------------- kernel 4: QK layernorm + RoPE + V convert -----------------
// Q,K -> [B,H,S,dh] bf16 (A / K^T operands);  V -> [B,H,dh,S] bf16 so the
// attention V fragment (contraction = key) is contiguous.
__global__ __launch_bounds__(256)
void qk_ln_rope_kernel(const float* __restrict__ qkv,
                       const float* __restrict__ qw, const float* __restrict__ kw,
                       __bf16* __restrict__ Qr, __bf16* __restrict__ Kr,
                       __bf16* __restrict__ Vt) {
    const int tok  = blockIdx.x;
    const int b    = tok / SS;
    const int spos = tok % SS;
    const float* q = qkv + (size_t)tok * 3 * DD;
    const float* k = q + DD;
    const float* v = k + DD;

    float sq = 0.f, sq2 = 0.f, sk = 0.f, sk2 = 0.f;
    for (int i = threadIdx.x; i < DD; i += 256) {
        float a = q[i]; sq += a; sq2 += a * a;
        float c = k[i]; sk += c; sk2 += c * c;
    }
    #pragma unroll
    for (int o = 16; o > 0; o >>= 1) {
        sq += __shfl_down(sq, o, 32);  sq2 += __shfl_down(sq2, o, 32);
        sk += __shfl_down(sk, o, 32);  sk2 += __shfl_down(sk2, o, 32);
    }
    __shared__ float red[4][8];
    const int wv = threadIdx.x >> 5, ln = threadIdx.x & 31;
    if (ln == 0) { red[0][wv]=sq; red[1][wv]=sq2; red[2][wv]=sk; red[3][wv]=sk2; }
    __syncthreads();
    float tq=0.f, tq2=0.f, tk=0.f, tk2=0.f;
    #pragma unroll
    for (int i = 0; i < 8; ++i) { tq+=red[0][i]; tq2+=red[1][i]; tk+=red[2][i]; tk2+=red[3][i]; }
    const float mq = tq / DD, rq = rsqrtf(tq2 / DD - mq * mq + LN_EPS);
    const float mk = tk / DD, rk = rsqrtf(tk2 / DD - mk * mk + LN_EPS);

    for (int i = threadIdx.x; i < DD; i += 256) {
        const int h = i >> 6, d = i & 63;
        const float qn = (q[i] - mq) * rq * qw[i];
        const float kn = (k[i] - mk) * rk * kw[i];
        const int pi = (d < 32) ? i + 32 : i - 32;          // rotate_half partner
        const float qp = (q[pi] - mq) * rq * qw[pi];
        const float kp = (k[pi] - mk) * rk * kw[pi];
        const float rotq = (d < 32) ? -qp : qp;
        const float rotk = (d < 32) ? -kp : kp;
        const int   fi   = d & 31;
        const float invf = __powf(10000.f, -(float)(2 * fi) / 64.f);
        const float ang  = (float)spos * invf;
        const float cs = __cosf(ang), sn = __sinf(ang);
        const size_t o = ((size_t)(b * HH + h) * SS + spos) * DH + d;
        Qr[o] = f2bf(qn * cs + rotq * sn);
        Kr[o] = f2bf(kn * cs + rotk * sn);
        Vt[((size_t)(b * HH + h) * DH + d) * SS + spos] = f2bf(v[i]);
    }
}

// -------------------- kernel 5: flash attention (WMMA) ---------------------
// grid = (S/128, H, B); 8 waves/block, each wave owns a 16-row q tile.
// Key loop: 32 keys/step, block-cooperative async K/V staging in LDS.
__global__ __launch_bounds__(256)
void attn_kernel(const __bf16* __restrict__ Qr, const __bf16* __restrict__ Kr,
                 const __bf16* __restrict__ Vt, const int* __restrict__ seq_id,
                 const unsigned char* __restrict__ maskv, __bf16* __restrict__ ctx) {
    const int b = blockIdx.z, h = blockIdx.y;
    const int wv   = threadIdx.x >> 5;
    const int lane = threadIdx.x & 31;
    const int l15  = lane & 15;
    const int kh   = lane >> 4;
    const int q0   = blockIdx.x * 128 + wv * 16;
    const size_t baseK = (size_t)(b * HH + h) * SS * DH;   // [S][dh]
    const size_t baseV = (size_t)(b * HH + h) * DH * SS;   // [dh][S]

    __shared__ alignas(16) __bf16 sK[32 * 64];   // [key][dh]  (dh innermost)
    __shared__ alignas(16) __bf16 sVt[64 * 32];  // [dh][key]  (key innermost)
    __shared__ alignas(16) __bf16 sP[8][16 * 32];

    // per-thread staging coordinates
    const int rk_ = threadIdx.x >> 3, ck = (threadIdx.x & 7) * 8;  // K tile 32x64
    const int rv  = threadIdx.x >> 2, cv = (threadIdx.x & 3) * 8;  // V tile 64x32
    const __bf16* gK = Kr + baseK + (size_t)rk_ * DH + ck;
    const __bf16* gV = Vt + baseV + (size_t)rv * SS + cv;
    __bf16* lK = &sK[rk_ * 64 + ck];
    __bf16* lV = &sVt[rv * 32 + cv];

    // Q fragments (dh=64 -> two K=32 chunks), A-matrix layout, global b128
    v16bf aq[2];
    #pragma unroll
    for (int c = 0; c < 2; ++c)
        aq[c] = load_frag_a(Qr + baseK + (size_t)(q0 + l15) * DH + c * 32, kh);

    int sidq[8];
    #pragma unroll
    for (int r = 0; r < 8; ++r) sidq[r] = seq_id[b * SS + q0 + r + 8 * kh];

    v8f acc[4];
    float mrun[8], lrun[8];
    #pragma unroll
    for (int t = 0; t < 4; ++t)
        #pragma unroll
        for (int r = 0; r < 8; ++r) acc[t][r] = 0.f;
    #pragma unroll
    for (int r = 0; r < 8; ++r) { mrun[r] = NEG_BIG; lrun[r] = 0.f; }
    const float scale = 0.125f;                 // 1/sqrt(64)

    for (int kj = 0; kj < SS; kj += 32) {
        // async DMA K/V tile staging, ASYNCcnt-tracked; prefetch next step
        async_ld_b128(gK + (size_t)kj * DH, lK);
        async_ld_b128(gV + kj,              lV);
        __builtin_prefetch(gK + (size_t)(kj + 32) * DH, 0, 1);
        __builtin_prefetch(gV + kj + 32, 0, 1);
        wait_async0();
        __syncthreads();

        // scores: two 16x16 tiles (32 keys), contraction dh=64 in two wmma each
        v8f sfr[2];
        #pragma unroll
        for (int t = 0; t < 2; ++t)
            #pragma unroll
            for (int r = 0; r < 8; ++r) sfr[t][r] = 0.f;
        #pragma unroll
        for (int nt = 0; nt < 2; ++nt) {
            const int n = nt * 16 + l15;        // key column (local)
            #pragma unroll
            for (int c = 0; c < 2; ++c) {
                const v16bf bk = load_frag_b(&sK[n * 64 + c * 32], kh); // K^T read: row-contig
                sfr[nt] = __builtin_amdgcn_wmma_f32_16x16x32_bf16(
                    false, aq[c], false, bk, (short)0, sfr[nt], false, false);
            }
        }

        // mask: allowed = !(same_seq && key_valid)   (faithful to reference)
        const int key0 = kj + l15, key1 = kj + 16 + l15;
        const int sid0 = seq_id[b * SS + key0], sid1 = seq_id[b * SS + key1];
        const int mk0 = maskv[b * SS + key0],  mk1 = maskv[b * SS + key1];

        float newmax[8];
        #pragma unroll
        for (int r = 0; r < 8; ++r) {
            float s0 = sfr[0][r] * scale;
            float s1 = sfr[1][r] * scale;
            if ((sidq[r] == sid0) && mk0) s0 = NEG_BIG;
            if ((sidq[r] == sid1) && mk1) s1 = NEG_BIG;
            sfr[0][r] = s0; sfr[1][r] = s1;
            float t = fmaxf(s0, s1);
            #pragma unroll
            for (int o = 1; o < 16; o <<= 1) t = fmaxf(t, __shfl_xor(t, o, 32));
            newmax[r] = fmaxf(mrun[r], t);      // half-wave = one 16-col row slice
        }
        #pragma unroll
        for (int r = 0; r < 8; ++r) {
            const float p0 = __expf(sfr[0][r] - newmax[r]);
            const float p1 = __expf(sfr[1][r] - newmax[r]);
            const float alpha = __expf(mrun[r] - newmax[r]);
            float rs = p0 + p1;
            #pragma unroll
            for (int o = 1; o < 16; o <<= 1) rs += __shfl_xor(rs, o, 32);
            lrun[r] = lrun[r] * alpha + rs;
            mrun[r] = newmax[r];
            const int mrow = r + 8 * kh;        // C-layout row for this lane
            sP[wv][mrow * 32 + l15]      = f2bf(p0);
            sP[wv][mrow * 32 + 16 + l15] = f2bf(p1);
            #pragma unroll
            for (int d = 0; d < 4; ++d) acc[d][r] *= alpha;
        }

        // P (16x32, A layout from LDS) @ V (32x64, key innermost -> contig B frag)
        const v16bf ap = load_frag_a(&sP[wv][l15 * 32], kh);
        #pragma unroll
        for (int d = 0; d < 4; ++d) {
            const v16bf bv = load_frag_b(&sVt[(d * 16 + l15) * 32], kh);
            acc[d] = __builtin_amdgcn_wmma_f32_16x16x32_bf16(
                false, ap, false, bv, (short)0, acc[d], false, false);
        }
        __syncthreads();
    }

    // ctx[b, q, h*64 + d]  (token-major for the output GEMM)
    #pragma unroll
    for (int d = 0; d < 4; ++d)
        #pragma unroll
        for (int r = 0; r < 8; ++r) {
            const int q = q0 + r + 8 * kh;
            const float val = (lrun[r] > 0.f) ? acc[d][r] / lrun[r] : 0.f;
            ctx[(size_t)(b * SS + q) * DD + h * DH + d * 16 + l15] = f2bf(val);
        }
}

// ---------------------------------------------------------------------------
extern "C" void kernel_launch(void* const* d_in, const int* in_sizes, int n_in,
                              void* d_out, int out_size, void* d_ws, size_t ws_size,
                              hipStream_t stream) {
    const float*         x      = (const float*)d_in[0];
    const int*           seq_id = (const int*)d_in[1];
    const unsigned char* maskv  = (const unsigned char*)d_in[2];
    const float*         ln_w   = (const float*)d_in[3];
    const float*         ln_b   = (const float*)d_in[4];
    const float*         w_qkv  = (const float*)d_in[5];
    const float*         q_ln_w = (const float*)d_in[6];
    const float*         k_ln_w = (const float*)d_in[7];
    const float*         w_out  = (const float*)d_in[8];
    float*               out    = (float*)d_out;
    char*                ws     = (char*)d_ws;

    // workspace layout (bytes, 256-aligned); total ~96 MiB
    constexpr size_t OFF_XN   = 0;                                    // 4096x1024 bf16
    constexpr size_t OFF_WQKV = OFF_XN   + (size_t)NT * DD * 2;       // 3072x1024 bf16 (W^T)
    constexpr size_t OFF_WOUT = OFF_WQKV + (size_t)DD * 3 * DD * 2;   // 1024x1024 bf16 (W^T)
    constexpr size_t OFF_QKV  = OFF_WOUT + (size_t)DD * DD * 2;       // 4096x3072 f32
    constexpr size_t OFF_QR   = OFF_QKV  + (size_t)NT * 3 * DD * 4;   // [B,H,S,dh] bf16
    constexpr size_t OFF_KR   = OFF_QR   + (size_t)NT * DD * 2;       // [B,H,S,dh] bf16
    constexpr size_t OFF_VT   = OFF_KR   + (size_t)NT * DD * 2;       // [B,H,dh,S] bf16
    constexpr size_t OFF_CTX  = OFF_VT   + (size_t)NT * DD * 2;       // 4096x1024 bf16

    __bf16* xn      = (__bf16*)(ws + OFF_XN);
    __bf16* wqkv_t  = (__bf16*)(ws + OFF_WQKV);
    __bf16* wout_t  = (__bf16*)(ws + OFF_WOUT);
    float*  qkv     = (float*) (ws + OFF_QKV);
    __bf16* Qr      = (__bf16*)(ws + OFF_QR);
    __bf16* Kr      = (__bf16*)(ws + OFF_KR);
    __bf16* Vt      = (__bf16*)(ws + OFF_VT);
    __bf16* ctx     = (__bf16*)(ws + OFF_CTX);

    // 1) layernorm(x) -> bf16
    ln_x_kernel<<<NT, 256, 0, stream>>>(x, ln_w, ln_b, xn);
    // 2) weights -> bf16, transposed to [N,K] so GEMM B-frags are contiguous
    transpose_cvt_kernel<<<dim3(3 * DD / 32, DD / 32), 256, 0, stream>>>(
        w_qkv, wqkv_t, DD, 3 * DD);
    transpose_cvt_kernel<<<dim3(DD / 32, DD / 32), 256, 0, stream>>>(
        w_out, wout_t, DD, DD);
    // 3) qkv = xn @ w_qkv   [4096,1024] x [1024,3072]
    gemm_bf16_kernel<<<dim3(3 * DD / 64, NT / 128), 256, 0, stream>>>(
        xn, wqkv_t, qkv, NT, 3 * DD, DD);
    // 4) qk layernorm + rope + v convert/transpose
    qk_ln_rope_kernel<<<NT, 256, 0, stream>>>(qkv, q_ln_w, k_ln_w, Qr, Kr, Vt);
    // 5) flash attention
    attn_kernel<<<dim3(SS / 128, HH, BB), 256, 0, stream>>>(
        Qr, Kr, Vt, seq_id, maskv, ctx);
    // 6) out = ctx @ w_out   [4096,1024] x [1024,1024]
    gemm_bf16_kernel<<<dim3(DD / 64, NT / 128), 256, 0, stream>>>(
        ctx, wout_t, out, NT, DD, DD);
}
